// StructureEncoder_59450937311581
// MI455X (gfx1250) — matrix operand order repaired and checked by reading, not compiled, not run on hardware
//
#include <hip/hip_runtime.h>

// ---------------------------------------------------------------------------
// StructureEncoder (3-layer GCN + input proj + mean pool) for MI455X (gfx1250)
// Dense GEMMs: v_wmma_f32_16x16x32_bf16, all bf16 conversion hoisted out of
// the GEMM inner loop (weights pre-transposed/converted once; activations
// mirrored in bf16 by the fused LN/ReLU kernel).
// ---------------------------------------------------------------------------

typedef __attribute__((ext_vector_type(16))) __bf16 v16bf;
typedef __attribute__((ext_vector_type(8)))  float  v8f;

static constexpr int HIDC = 128;          // NODE_DIM == HID == 128

__device__ __forceinline__ unsigned short f32_to_bf16_rne(float f) {
    union { float f; unsigned u; } c; c.f = f;
    unsigned u = c.u;
    u += 0x7FFFu + ((u >> 16) & 1u);      // round-to-nearest-even
    return (unsigned short)(u >> 16);
}

// ---------------------------------------------------------------------------
// Y[nrows,128] = Xbf[nrows,128] @ W  (+ bias), W given as bf16 transpose
// WTbf[n*128+k].  One block = 16 output rows; 8 waves, one 16-col tile each.
// Inner loop per wave: 4 x (2 A-loads b128 + 2 B-loads b128 + 1 wmma).
// ---------------------------------------------------------------------------
__global__ __launch_bounds__(256)
void gemm128_bf16wmma(const unsigned short* __restrict__ Xbf,
                      const unsigned short* __restrict__ WTbf,
                      const float* __restrict__ bias,
                      float* __restrict__ Y,
                      unsigned short* __restrict__ Ybf,   // optional bf16 mirror
                      int nrows) {
    const int t       = threadIdx.x;
    const int wave    = t >> 5;           // 0..7 -> 16-col tile
    const int lane    = t & 31;
    const int m       = lane & 15;
    const int half    = lane >> 4;        // 0: lanes 0-15, 1: lanes 16-31
    const int colbase = wave * 16;
    const int rowbase = blockIdx.x * 16;

    int arow = rowbase + m;
    if (arow >= nrows) arow = nrows - 1;  // clamp (N is a multiple of 16 here)
    const unsigned short* xrow = Xbf  + (size_t)arow * HIDC;
    const unsigned short* wcol = WTbf + (size_t)(colbase + m) * HIDC;

    v8f acc = {};
    #pragma unroll
    for (int kb = 0; kb < HIDC; kb += 32) {
        // A 16x32 bf16: lane half h holds K in {kb+8h..+7} and {kb+16+8h..+7}
        union { v16bf v; uint4 q[2]; } a;
        a.q[0] = *(const uint4*)(xrow + kb + half * 8);
        a.q[1] = *(const uint4*)(xrow + kb + 16 + half * 8);
        // B 32x16 bf16: lane half h holds K in {kb+8h..+7} and {kb+16+8h..+7}, col n
        union { v16bf v; uint4 q[2]; } b;
        b.q[0] = *(const uint4*)(wcol + kb + half * 8);
        b.q[1] = *(const uint4*)(wcol + kb + 16 + half * 8);

        acc = __builtin_amdgcn_wmma_f32_16x16x32_bf16(
            /*neg_a=*/false, a.v, /*neg_b=*/false, b.v,
            /*c_mod=*/(short)0, acc, /*reuse_a=*/false, /*reuse_b=*/false);
    }

    // D layout: VGPR r -> row M = half*8 + r, col N = lane&15
    const int col  = colbase + m;
    const float bv = bias ? bias[col] : 0.0f;
    #pragma unroll
    for (int r = 0; r < 8; ++r) {
        int row = rowbase + half * 8 + r;
        if (row < nrows) {
            float v = acc[r] + bv;
            Y[(size_t)row * HIDC + col] = v;
            if (Ybf) Ybf[(size_t)row * HIDC + col] = f32_to_bf16_rne(v);
        }
    }
}

// ---------------------------------------------------------------------------
// Transpose + convert nmats 128x128 f32 matrices to bf16:  WT[n][k] = W[k][n]
// ---------------------------------------------------------------------------
__global__ __launch_bounds__(256)
void wconv_k(const float* __restrict__ W, unsigned short* __restrict__ WT, int nmats) {
    int t = blockIdx.x * blockDim.x + threadIdx.x;
    int total = nmats * HIDC * HIDC;
    if (t >= total) return;
    int mat = t >> 14;                 // / 16384
    int idx = t & 16383;
    int k = idx >> 7, n = idx & 127;
    WT[(size_t)mat * HIDC * HIDC + n * HIDC + k] = f32_to_bf16_rne(W[t]);
}

// f32 -> bf16 flat convert (for the initial node features)
__global__ __launch_bounds__(256)
void conv_bf_k(const float* __restrict__ in, unsigned short* __restrict__ out, int n) {
    int t = blockIdx.x * blockDim.x + threadIdx.x;
    if (t < n) out[t] = f32_to_bf16_rne(in[t]);
}

// ---------------------------------------------------------------------------
// deg[dst] += 1 over all edges
// ---------------------------------------------------------------------------
__global__ __launch_bounds__(256)
void degree_k(const int* __restrict__ dst, float* __restrict__ deg, int E) {
    int t = blockIdx.x * blockDim.x + threadIdx.x;
    if (t < E) atomicAdd(&deg[dst[t]], 1.0f);
}

// deg -> dinv_sqrt (in place) and dinv, with self-loop (+1)
__global__ __launch_bounds__(256)
void dinv_k(float* __restrict__ deg_dis, float* __restrict__ dinv, int n) {
    int t = blockIdx.x * blockDim.x + threadIdx.x;
    if (t >= n) return;
    float d = deg_dis[t] + 1.0f;
    deg_dis[t] = rsqrtf(d);
    dinv[t]    = 1.0f / d;
}

// ---------------------------------------------------------------------------
// agg[dst] += xw[src] * dinv_sqrt[src]*dinv_sqrt[dst]  (one wave per edge)
// ---------------------------------------------------------------------------
__global__ __launch_bounds__(256)
void scatter_k(const int* __restrict__ src, const int* __restrict__ dst,
               const float* __restrict__ dis, const float* __restrict__ xw,
               float* __restrict__ agg, int E) {
    int wid  = (blockIdx.x * blockDim.x + threadIdx.x) >> 5;
    if (wid >= E) return;
    int lane = threadIdx.x & 31;
    int s = src[wid], d = dst[wid];
    float en = dis[s] * dis[d];
    float4 v = *(const float4*)(xw + (size_t)s * HIDC + lane * 4);
    float* p = agg + (size_t)d * HIDC + lane * 4;
    atomicAdd(p + 0, v.x * en);
    atomicAdd(p + 1, v.y * en);
    atomicAdd(p + 2, v.z * en);
    atomicAdd(p + 3, v.w * en);
}

// ---------------------------------------------------------------------------
// x = relu(LN(agg + xw*dinv + b)) [+ res]; also writes bf16 mirror for the
// next layer's WMMA A operand.  One wave per node row.
// ---------------------------------------------------------------------------
__global__ __launch_bounds__(256)
void update_k(const float* __restrict__ agg, const float* __restrict__ xw,
              const float* __restrict__ dinv, const float* __restrict__ bias,
              const float* __restrict__ g, const float* __restrict__ bln,
              float* __restrict__ x, unsigned short* __restrict__ xbf,
              int n, int add_res) {
    int wid  = blockIdx.x * (blockDim.x >> 5) + (threadIdx.x >> 5);
    int lane = threadIdx.x & 31;
    if (wid >= n) return;
    size_t base = (size_t)wid * HIDC + lane * 4;
    float di = dinv[wid];
    float4 va  = *(const float4*)(agg + base);
    float4 vw  = *(const float4*)(xw + base);
    float4 rs4 = *(const float4*)(x + base);
    float4 bb  = *(const float4*)(bias + lane * 4);

    float v0 = va.x + vw.x * di + bb.x;
    float v1 = va.y + vw.y * di + bb.y;
    float v2 = va.z + vw.z * di + bb.z;
    float v3 = va.w + vw.w * di + bb.w;

    float s = v0 + v1 + v2 + v3;
    #pragma unroll
    for (int off = 16; off > 0; off >>= 1) s += __shfl_xor(s, off, 32);
    float mu = s * (1.0f / 128.0f);

    float d0 = v0 - mu, d1 = v1 - mu, d2 = v2 - mu, d3 = v3 - mu;
    float q = d0 * d0 + d1 * d1 + d2 * d2 + d3 * d3;
    #pragma unroll
    for (int off = 16; off > 0; off >>= 1) q += __shfl_xor(q, off, 32);
    float rstd = rsqrtf(q * (1.0f / 128.0f) + 1e-5f);

    float4 gg = *(const float4*)(g + lane * 4);
    float4 lb = *(const float4*)(bln + lane * 4);
    float y0 = fmaxf(d0 * rstd * gg.x + lb.x, 0.0f);
    float y1 = fmaxf(d1 * rstd * gg.y + lb.y, 0.0f);
    float y2 = fmaxf(d2 * rstd * gg.z + lb.z, 0.0f);
    float y3 = fmaxf(d3 * rstd * gg.w + lb.w, 0.0f);
    if (add_res) { y0 += rs4.x; y1 += rs4.y; y2 += rs4.z; y3 += rs4.w; }

    float4 out; out.x = y0; out.y = y1; out.z = y2; out.w = y3;
    *(float4*)(x + base) = out;

    ushort4 ob;
    ob.x = f32_to_bf16_rne(y0); ob.y = f32_to_bf16_rne(y1);
    ob.z = f32_to_bf16_rne(y2); ob.w = f32_to_bf16_rne(y3);
    *(ushort4*)(xbf + base) = ob;
}

// ---------------------------------------------------------------------------
// Mean pool: plan[b] += x[node], counts[b] += 1   (one wave per node)
// ---------------------------------------------------------------------------
__global__ __launch_bounds__(256)
void pool_k(const float* __restrict__ x, const int* __restrict__ bidx,
            float* __restrict__ plan, float* __restrict__ counts, int n) {
    int wid  = blockIdx.x * (blockDim.x >> 5) + (threadIdx.x >> 5);
    int lane = threadIdx.x & 31;
    if (wid >= n) return;
    int b = bidx[wid];
    if (lane == 0) atomicAdd(&counts[b], 1.0f);
    float4 v = *(const float4*)(x + (size_t)wid * HIDC + lane * 4);
    float* p = plan + (size_t)b * HIDC + lane * 4;
    atomicAdd(p + 0, v.x);
    atomicAdd(p + 1, v.y);
    atomicAdd(p + 2, v.z);
    atomicAdd(p + 3, v.w);
}

__global__ __launch_bounds__(256)
void pool_fin_k(float* __restrict__ plan, const float* __restrict__ counts, int total) {
    int t = blockIdx.x * blockDim.x + threadIdx.x;
    if (t < total) plan[t] /= fmaxf(counts[t >> 7], 1.0f);
}

// ---------------------------------------------------------------------------
extern "C" void kernel_launch(void* const* d_in, const int* in_sizes, int n_in,
                              void* d_out, int out_size, void* d_ws, size_t ws_size,
                              hipStream_t stream) {
    const float* nodes     = (const float*)d_in[0];
    const int*   edges     = (const int*)d_in[1];
    // d_in[2] = edge_types: unused by the reference
    const int*   batch_idx = (const int*)d_in[3];
    const float* w_in      = (const float*)d_in[4];
    const float* b_in      = (const float*)d_in[5];
    const float* gcn_w     = (const float*)d_in[6];
    const float* gcn_b     = (const float*)d_in[7];
    const float* ln_g      = (const float*)d_in[8];
    const float* ln_b      = (const float*)d_in[9];

    const int N = in_sizes[0] / HIDC;
    const int E = in_sizes[1] / 2;
    const int L = in_sizes[6] / (HIDC * HIDC);
    const int B = out_size / HIDC - N;

    const int* srcI = edges;
    const int* dstI = edges + E;

    // Output layout: x [N,128] then plan [B,128]
    float* x    = (float*)d_out;
    float* plan = x + (size_t)N * HIDC;

    // Workspace layout (f32 arrays first; all offsets stay 16B-aligned)
    float* xw     = (float*)d_ws;
    float* agg    = xw + (size_t)N * HIDC;
    float* dis    = agg + (size_t)N * HIDC;  // deg, then rsqrt(deg+1) in place
    float* dinv   = dis + N;
    float* counts = dinv + N;
    unsigned short* xbf  = (unsigned short*)(counts + B);            // [N,128] bf16
    unsigned short* wtbf = xbf + (size_t)N * HIDC;                   // (1+L) x 128x128 bf16^T
    size_t need = ((size_t)N * HIDC * 2 + (size_t)N * 2 + (size_t)B) * sizeof(float)
                + ((size_t)N * HIDC + (size_t)(1 + L) * HIDC * HIDC) * sizeof(unsigned short);
    if (ws_size < need) return;

    // 0) pre-convert weights (transposed bf16) and node features (bf16)
    wconv_k<<<(1 * HIDC * HIDC + 255) / 256, 256, 0, stream>>>(w_in, wtbf, 1);
    wconv_k<<<(L * HIDC * HIDC + 255) / 256, 256, 0, stream>>>(gcn_w, wtbf + (size_t)HIDC * HIDC, L);
    conv_bf_k<<<(N * HIDC + 255) / 256, 256, 0, stream>>>(nodes, xbf, N * HIDC);

    // 1) degree + normalization terms
    hipMemsetAsync(dis, 0, (size_t)N * sizeof(float), stream);
    degree_k<<<(E + 255) / 256, 256, 0, stream>>>(dstI, dis, E);
    dinv_k<<<(N + 255) / 256, 256, 0, stream>>>(dis, dinv, N);

    // 2) input projection: x = nodes @ w_in + b_in  (writes f32 + bf16 mirror)
    gemm128_bf16wmma<<<(N + 15) / 16, 256, 0, stream>>>(xbf, wtbf, b_in, x, xbf, N);

    // 3) GCN layers
    const long long edgeThreads = (long long)E * 32;
    const unsigned scatterBlocks = (unsigned)((edgeThreads + 255) / 256);
    for (int i = 0; i < L; ++i) {
        gemm128_bf16wmma<<<(N + 15) / 16, 256, 0, stream>>>(
            xbf, wtbf + (size_t)(i + 1) * HIDC * HIDC, nullptr, xw, nullptr, N);
        hipMemsetAsync(agg, 0, (size_t)N * HIDC * sizeof(float), stream);
        scatter_k<<<scatterBlocks, 256, 0, stream>>>(srcI, dstI, dis, xw, agg, E);
        update_k<<<(N + 7) / 8, 256, 0, stream>>>(
            agg, xw, dinv, gcn_b + (size_t)i * HIDC,
            ln_g + (size_t)i * HIDC, ln_b + (size_t)i * HIDC, x, xbf, N, i > 0);
    }

    // 4) global mean pool
    hipMemsetAsync(plan, 0, (size_t)B * HIDC * sizeof(float), stream);
    hipMemsetAsync(counts, 0, (size_t)B * sizeof(float), stream);
    pool_k<<<(N + 7) / 8, 256, 0, stream>>>(x, batch_idx, plan, counts, N);
    pool_fin_k<<<(B * HIDC + 255) / 256, 256, 0, stream>>>(plan, counts, B * HIDC);
}
